// BVPVelocityLoss_44092134261355
// MI455X (gfx1250) — matrix operand order repaired and checked by reading, not compiled, not run on hardware
//
#include <hip/hip_runtime.h>
#include <hip/hip_bf16.h>
#include <math.h>

// ---------------------------------------------------------------------------
// BVP velocity loss for MI455X (gfx1250, wave32).
//   pearson + peak(count/value/freq) + derivative-cosine losses.
// Band-limited periodogram computed as a bf16 WMMA GEMM against a DFT basis
// resident in L2 (60 MB < 192 MB L2); everything else is one streaming pass.
// ---------------------------------------------------------------------------

typedef __attribute__((ext_vector_type(16))) __bf16 v16bf;
typedef __attribute__((ext_vector_type(8)))  float  v8f;

#define B_ROWS 2048
#define T_LEN  16384          // power of two -> nfft == T_LEN
#define FS_HZ  30.0f
#define KMIN   410            // ceil(0.75*16384/30)
#define NBINS  956            // bins 410..1365 inclusive
#define NCOLS  1912           // 2*NBINS (cos,sin interleaved)
#define NCOLP  1920           // padded to 120 tiles of 16
#define NT_PER_WAVE 15        // 8 waves * 15 tiles * 16 cols = 1920
#define NWAVES 8

// ---------------------------------------------------------------------------
// block-wide sum over 256 threads (8 wave32s)
// ---------------------------------------------------------------------------
__device__ __forceinline__ float blockReduceSum256(float v) {
    __shared__ float wsum[NWAVES];
    const int lane = threadIdx.x & 31;
    const int w    = threadIdx.x >> 5;
    #pragma unroll
    for (int o = 16; o > 0; o >>= 1) v += __shfl_down(v, o, 32);
    __syncthreads();                       // protect wsum across repeated calls
    if (lane == 0) wsum[w] = v;
    __syncthreads();
    v = (threadIdx.x < NWAVES) ? wsum[threadIdx.x] : 0.0f;
    if (w == 0) {
        #pragma unroll
        for (int o = NWAVES / 2; o > 0; o >>= 1) v += __shfl_down(v, o, 32);
    }
    return v;                              // valid on thread 0
}

// ---------------------------------------------------------------------------
// np.gradient helpers operating on an LDS tile with 2-element halo.
// li = local index into tile, g = global index in [0, T_LEN)
// ---------------------------------------------------------------------------
__device__ __forceinline__ float grad1(const float* x, int li, int g) {
    if (g == 0)         return x[li + 1] - x[li];
    if (g == T_LEN - 1) return x[li] - x[li - 1];
    return 0.5f * (x[li + 1] - x[li - 1]);
}
__device__ __forceinline__ float grad2(const float* x, int li, int g) {
    if (g == 0)         return grad1(x, li + 1, 1) - grad1(x, li, 0);
    if (g == T_LEN - 1) return grad1(x, li, g) - grad1(x, li - 1, g - 1);
    return 0.5f * (grad1(x, li + 1, g + 1) - grad1(x, li - 1, g - 1));
}

// ---------------------------------------------------------------------------
// Kernel 0: DFT basis, bf16, row n of basisT holds cos/sin(2*pi*kbin*t/16384).
// Exact integer phase reduction: (kbin*t) mod 16384 keeps f32 trig accurate.
// ---------------------------------------------------------------------------
__global__ __launch_bounds__(256) void k_basis(__bf16* __restrict__ basisT) {
    const int col = blockIdx.x;                    // [0, NCOLP)
    __bf16* dst = basisT + ((size_t)col << 14);    // *16384
    if (col >= NCOLS) {                            // padded columns -> zero
        for (int n = threadIdx.x; n < T_LEN; n += 256) dst[n] = (__bf16)0.0f;
        return;
    }
    const unsigned kbin = (unsigned)(KMIN + (col >> 1));
    const bool is_sin = (col & 1) != 0;
    const float w = 6.283185307179586f / 16384.0f;
    for (int n = threadIdx.x; n < T_LEN; n += 256) {
        unsigned m = (kbin * (unsigned)n) & 16383u;
        float sv, cv;
        __sincosf(w * (float)m, &sv, &cv);
        dst[n] = (__bf16)(is_sin ? sv : cv);
    }
}

// ---------------------------------------------------------------------------
// Kernel 1: fused per-row statistics (one HBM pass over both inputs).
// 17 reductions per row: pearson sums, deriv-cosine sums, peak counts/sums.
// ---------------------------------------------------------------------------
__global__ __launch_bounds__(256) void k_rowstats(const float* __restrict__ P,
                                                  const float* __restrict__ Tg,
                                                  float* __restrict__ stats) {
    const int row = blockIdx.x;
    const int tid = threadIdx.x;
    const float* p = P  + (size_t)row * T_LEN;
    const float* t = Tg + (size_t)row * T_LEN;

    __shared__ float sp[2052];   // 2048 + 2 halo each side
    __shared__ float st[2052];

    float a[17];
    #pragma unroll
    for (int q = 0; q < 17; q++) a[q] = 0.0f;

    for (int tile = 0; tile < T_LEN / 2048; ++tile) {
        const int t0 = tile * 2048;
        if (t0 + 2048 < T_LEN) {                       // gfx1250 prefetch path
            __builtin_prefetch(p + t0 + 2048 + tid * 8, 0, 1);
            __builtin_prefetch(t + t0 + 2048 + tid * 8, 0, 1);
        }
        __syncthreads();
        for (int j = tid; j < 2052; j += 256) {
            int g = t0 - 2 + j;
            g = g < 0 ? 0 : (g > T_LEN - 1 ? T_LEN - 1 : g);
            sp[j] = p[g];
            st[j] = t[g];
        }
        __syncthreads();
        #pragma unroll 2
        for (int k = 0; k < 2048 / 256; k++) {
            const int li = 2 + tid + k * 256;
            const int g  = t0 + tid + k * 256;
            const float pv = sp[li], tv = st[li];
            // pearson raw sums (standardization is a no-op for r)
            a[0] += pv; a[1] += tv; a[2] += pv * pv; a[3] += tv * tv; a[4] += pv * tv;
            // derivative cosine terms
            const float g1p = grad1(sp, li, g), g1t = grad1(st, li, g);
            a[5] += g1p * g1t; a[6] += g1p * g1p; a[7] += g1t * g1t;
            const float g2p = grad2(sp, li, g), g2t = grad2(st, li, g);
            a[8] += g2p * g2t; a[9] += g2p * g2p; a[10] += g2t * g2t;
            // strict interior peaks / troughs
            const bool inb = (g >= 1) && (g <= T_LEN - 2);
            const float pm1 = sp[li - 1], pp1 = sp[li + 1];
            const float tm1 = st[li - 1], tp1 = st[li + 1];
            const bool pkp = inb && (pv > pm1) && (pv > pp1);
            const bool mnp = inb && (pv < pm1) && (pv < pp1);
            const bool pkt = inb && (tv > tm1) && (tv > tp1);
            const bool mnt = inb && (tv < tm1) && (tv < tp1);
            a[11] += pkt ? 1.0f : 0.0f;
            a[12] += pkp ? 1.0f : 0.0f;
            a[13] += pkp ? pv   : 0.0f;
            a[14] += mnt ? 1.0f : 0.0f;
            a[15] += mnp ? 1.0f : 0.0f;
            a[16] += mnp ? pv   : 0.0f;
        }
    }
    for (int q = 0; q < 17; q++) {
        const float s = blockReduceSum256(a[q]);
        if (tid == 0) stats[row * 20 + q] = s;
    }
}

// ---------------------------------------------------------------------------
// Kernel 2: band-limited DFT as bf16 WMMA GEMM + per-row periodogram argmax.
// Block = 8 wave32s, owns rows [16b, 16b+16) of the 4096-row virtual matrix
// (rows 0..2047 = predictions, 2048..4095 = targets), sweeps all 1920 padded
// columns (15 N-tiles per wave). A is read once from HBM; basis streams from
// L2. A-tile staging is ping-ponged (1 barrier/iter) and B fragments are
// double-buffered so tile j+1's loads fly during WMMA j (partial loadcnt waits).
// ---------------------------------------------------------------------------
__global__ __launch_bounds__(256) void k_dft_argmax(const float* __restrict__ P,
                                                    const float* __restrict__ Tg,
                                                    const __bf16* __restrict__ basisT,
                                                    float* __restrict__ fpeak) {
    __shared__ __bf16 aS[2][16 * 32];            // ping-pong 16x32 bf16 tiles
    __shared__ unsigned long long best[16];      // (power_bits<<32)|bin per row

    const int tid  = threadIdx.x;
    const int lane = tid & 31;
    const int wave = tid >> 5;
    const int hi   = (lane >> 4) & 1;            // lane half (K-half select)
    const int l15  = lane & 15;

    if (tid < 16) best[tid] = 0ull;

    // A-tile staging assignment: 512 f32 elements, 2 per thread
    const int e  = tid * 2;
    const int am = e >> 5;                       // staged row 0..15
    const int ak = e & 31;                       // staged K offset 0..30 (even)
    const int vr_fill = blockIdx.x * 16 + am;
    const float* src = (vr_fill < B_ROWS)
                       ? (P  + (size_t)vr_fill * T_LEN)
                       : (Tg + (size_t)(vr_fill - B_ROWS) * T_LEN);

    // Per-lane 32-bit byte offsets into the basis (max ~63 MB < 2^31) so the
    // backend can use scalar-base + vector-offset (GVS) addressing.
    unsigned boff[NT_PER_WAVE];
    #pragma unroll
    for (int j = 0; j < NT_PER_WAVE; j++) {
        const unsigned col = (unsigned)((wave * NT_PER_WAVE + j) * 16 + l15);
        boff[j] = (col << 15) + ((unsigned)hi << 5);   // col*16384*2 + hi*32
    }

    v8f acc[NT_PER_WAVE];
    const v8f vzero = {0.f, 0.f, 0.f, 0.f, 0.f, 0.f, 0.f, 0.f};
    #pragma unroll
    for (int j = 0; j < NT_PER_WAVE; j++) acc[j] = vzero;

    union Frag { v16bf v; uint4 q[2]; };

    // stage tile 0
    aS[0][am * 32 + ak]     = (__bf16)src[ak];
    aS[0][am * 32 + ak + 1] = (__bf16)src[ak + 1];
    __syncthreads();

    for (int k0 = 0, it = 0; k0 < T_LEN; k0 += 32, ++it) {
        const int cur = it & 1;
        const int nxt = cur ^ 1;

        // A fragment (16x32 bf16): per lane two 8-elem K-chunks
        //   elems[0..7]  = K 16*hi+{0..7};  elems[8..15] = K 16*hi+8+{0..7}
        Frag af;
        const __bf16* abase = aS[cur] + l15 * 32;
        af.q[0] = *(const uint4*)(abase + (hi << 4));
        af.q[1] = *(const uint4*)(abase + 8 + (hi << 4));

        // issue next A-tile f32 loads early; latency hides behind 15 WMMAs
        const int kn = k0 + 32;
        float f0 = 0.0f, f1 = 0.0f;
        if (kn < T_LEN) {
            f0 = src[kn + ak];
            f1 = src[kn + ak + 1];
            if (kn + 32 < T_LEN) __builtin_prefetch(src + kn + 32 + ak, 0, 1);
        }

        // pipelined B loads + WMMA: load j+1 while multiplying j
        const char* bbase = (const char*)basisT + ((size_t)k0 * 2);
        Frag bb[2];
        bb[0].q[0] = ((const uint4*)(bbase + boff[0]))[0];
        bb[0].q[1] = ((const uint4*)(bbase + boff[0]))[1];
        #pragma unroll
        for (int j = 0; j < NT_PER_WAVE; j++) {
            if (j + 1 < NT_PER_WAVE) {
                const uint4* np = (const uint4*)(bbase + boff[j + 1]);
                bb[(j + 1) & 1].q[0] = np[0];
                bb[(j + 1) & 1].q[1] = np[1];
            }
            acc[j] = __builtin_amdgcn_wmma_f32_16x16x32_bf16(
                         false, af.v, false, bb[j & 1].v,
                         (short)0, acc[j], false, false);
        }

        // stage next tile into the other buffer (readers of it synced already)
        if (kn < T_LEN) {
            aS[nxt][am * 32 + ak]     = (__bf16)f0;
            aS[nxt][am * 32 + ak + 1] = (__bf16)f1;
        }
        __syncthreads();
    }

    // Epilogue: C[m][n] in lane (n = l15, m = 8*hi + r). cos/sin of a bin sit
    // in adjacent lanes; pair via shuffle, argmax via monotone u64 LDS atomic.
    #pragma unroll
    for (int j = 0; j < NT_PER_WAVE; j++) {
        const int col = (wave * NT_PER_WAVE + j) * 16 + l15;
        #pragma unroll
        for (int r = 0; r < 8; r++) {
            const float re = acc[j][r];
            const float im = __shfl_xor(re, 1, 32);   // partner lane (sin col)
            if (!(lane & 1)) {
                const int bin = col >> 1;
                if (bin < NBINS) {
                    const float pw = re * re + im * im;   // >= 0: bits monotone
                    const unsigned long long key =
                        ((unsigned long long)__float_as_uint(pw) << 32) |
                        (unsigned long long)(unsigned)bin;
                    atomicMax(&best[(hi << 3) + r], key); // ds_max_u64
                }
            }
        }
    }
    __syncthreads();
    if (tid < 16) {
        const unsigned bin = (unsigned)(best[tid] & 0xffffffffull);
        fpeak[blockIdx.x * 16 + tid] =
            (float)(KMIN + (int)bin) * (FS_HZ / 16384.0f);
    }
}

// ---------------------------------------------------------------------------
// Kernel 3: combine per-row stats + peak frequencies into the scalar loss.
// ---------------------------------------------------------------------------
__global__ __launch_bounds__(256) void k_finalize(const float* __restrict__ stats,
                                                  const float* __restrict__ fpeak,
                                                  float* __restrict__ out) {
    const int tid = threadIdx.x;
    float s_pe = 0.0f, s_pk = 0.0f, s_c = 0.0f;
    const float Nf = (float)T_LEN;
    for (int r = tid; r < B_ROWS; r += 256) {
        const float* s = stats + r * 20;
        const float sp = s[0], st_ = s[1], spp = s[2], stt = s[3], spt = s[4];
        const float num = Nf * spt - sp * st_;
        const float den = sqrtf((Nf * spp - sp * sp) * (Nf * stt - st_ * st_));
        s_pe += 1.0f - num / den;                                   // pearson
        s_c  += s[5] * rsqrtf(s[6] * s[7])                          // c1
              + s[8] * rsqrtf(s[9] * s[10]);                        // c2
        const float cnt_d  = fabsf(s[11] - s[12]);
        const float val_d  = fabsf(1.0f - s[13] / s[12]);
        const float ncnt_d = fabsf(s[14] - s[15]);
        const float nval_d = fabsf(1.0f - s[16] / s[15]);
        const float fd     = fabsf(fpeak[B_ROWS + r] - fpeak[r]);
        s_pk += 0.5f * (cnt_d + ncnt_d + val_d + nval_d) + fd;
    }
    s_pe = blockReduceSum256(s_pe);
    s_pk = blockReduceSum256(s_pk);
    s_c  = blockReduceSum256(s_c);
    if (tid == 0)
        out[0] = s_pe / (float)B_ROWS + s_pk / (float)B_ROWS
               + 2.0f - s_c / (float)B_ROWS;
}

// ---------------------------------------------------------------------------
// Host launcher. Workspace layout:
//   [0, 62914560)        bf16 basisT [1920][16384]           (~60 MB, L2-hot)
//   [+0, +163840)        f32 stats   [2048][20]
//   [+,  +16384)         f32 fpeak   [4096]  (pred rows, then target rows)
// ---------------------------------------------------------------------------
extern "C" void kernel_launch(void* const* d_in, const int* in_sizes, int n_in,
                              void* d_out, int out_size, void* d_ws, size_t ws_size,
                              hipStream_t stream) {
    (void)in_sizes; (void)n_in; (void)out_size; (void)ws_size;
    const float* P  = (const float*)d_in[0];
    const float* Tg = (const float*)d_in[1];
    float* out = (float*)d_out;

    char* ws = (char*)d_ws;
    const size_t BASIS_BYTES = (size_t)NCOLP * T_LEN * sizeof(__bf16); // 62,914,560
    const size_t STATS_BYTES = (size_t)B_ROWS * 20 * sizeof(float);    // 163,840
    __bf16* basisT = (__bf16*)ws;
    float*  stats  = (float*)(ws + BASIS_BYTES);
    float*  fpeak  = (float*)(ws + BASIS_BYTES + STATS_BYTES);

    k_basis     <<<NCOLP,         256, 0, stream>>>(basisT);
    k_rowstats  <<<B_ROWS,        256, 0, stream>>>(P, Tg, stats);
    k_dft_argmax<<<(2*B_ROWS)/16, 256, 0, stream>>>(P, Tg, basisT, fpeak);
    k_finalize  <<<1,             256, 0, stream>>>(stats, fpeak, out);
}
